// GATModel_71777493450787
// MI455X (gfx1250) — matrix operand order repaired and checked by reading, not compiled, not run on hardware
//
#include <hip/hip_runtime.h>
#include <hip/hip_bf16.h>

// ---------------- problem constants (from reference) ----------------
#define NN    50000          // nodes
#define NE    800000         // edges
#define FDIM  128            // F_IN == H*C
#define NH    4              // heads
#define NC    32             // per-head channels
#define NED   16             // edge_dim
#define NL    3              // layers
#define NEG_SLOPE 0.2f
#define EF    (NE + NN)      // edges + self loops

typedef __attribute__((ext_vector_type(16))) __bf16 v16bf;
typedef __attribute__((ext_vector_type(8)))  float  v8f;
typedef __attribute__((ext_vector_type(4)))  unsigned u32x4;

// monotone encode/decode so unsigned atomicMax == float max
__device__ __forceinline__ unsigned fenc(float f) {
    unsigned u = __float_as_uint(f);
    return (u & 0x80000000u) ? ~u : (u | 0x80000000u);
}
__device__ __forceinline__ float fdec(unsigned u) {
    u = (u & 0x80000000u) ? (u & 0x7FFFFFFFu) : ~u;
    return __uint_as_float(u);
}

// ---------------- generic fill ----------------
__global__ void k_fill_f32(float* __restrict__ p, float v, int n) {
    int i = blockIdx.x * blockDim.x + threadIdx.x;
    if (i < n) p[i] = v;
}

// ---------------- prep kernels ----------------
__global__ void k_x_to_bf(const float* __restrict__ x, __bf16* __restrict__ hbf, int n) {
    int i = blockIdx.x * blockDim.x + threadIdx.x;
    if (i < n) hbf[i] = (__bf16)x[i];
}

// WT[l][n][k] = bf16(W_src[l][k][n])  (transpose so B-fragment loads are contiguous)
__global__ void k_wt(const float* __restrict__ Wsrc, __bf16* __restrict__ WT) {
    int i = blockIdx.x * blockDim.x + threadIdx.x;
    if (i >= NL * FDIM * FDIM) return;
    int l = i / (FDIM * FDIM);
    int r = i % (FDIM * FDIM);
    int n = r / FDIM, k = r % FDIM;
    WT[i] = (__bf16)Wsrc[(size_t)l * FDIM * FDIM + (size_t)k * FDIM + n];
}

// v_edge[l][d][h] = sum_c W_edge[l][d][h*C+c] * att_edge[l][h][c]
__global__ void k_vedge(const float* __restrict__ Wedge, const float* __restrict__ att_edge,
                        float* __restrict__ vedge) {
    int i = blockIdx.x * blockDim.x + threadIdx.x;
    if (i >= NL * NED * NH) return;
    int l = i / (NED * NH);
    int r = i % (NED * NH);
    int d = r / NH, h = r % NH;
    float s = 0.f;
    const float* w = Wedge + (size_t)l * NED * FDIM + (size_t)d * FDIM + h * NC;
    const float* a = att_edge + (size_t)l * NH * NC + h * NC;
    for (int c = 0; c < NC; ++c) s += w[c] * a[c];
    vedge[i] = s;
}

// degree + per-node sum of incoming edge_attr (for self-loop fill_value='mean')
__global__ void k_deg_loop(const int* __restrict__ ei, const float* __restrict__ eattr,
                           float* __restrict__ deg, float* __restrict__ loopsum) {
    int e = blockIdx.x * blockDim.x + threadIdx.x;
    if (e >= NE) return;
    int d = ei[NE + e];
    atomicAdd(deg + d, 1.0f);
    const float* ea = eattr + (size_t)e * NED;
    float* ls = loopsum + (size_t)d * NED;
    for (int j = 0; j < NED; ++j) atomicAdd(ls + j, ea[j]);
}

__global__ void k_loop_div(const float* __restrict__ loopsum, const float* __restrict__ deg,
                           float* __restrict__ loop_attr) {
    int i = blockIdx.x * blockDim.x + threadIdx.x;
    if (i >= NN * NED) return;
    loop_attr[i] = loopsum[i] / fmaxf(deg[i / NED], 1.0f);
}

// ---------------- WMMA GEMM: hp[N,128] = hbf[N,128] @ W (via WT bf16) ----------------
// one wave -> one 16x16 tile, K=128 in 4 steps of v_wmma_f32_16x16x32_bf16
__global__ __launch_bounds__(128) void k_gemm_hp(const __bf16* __restrict__ hbf,
                                                 const __bf16* __restrict__ WT,
                                                 float* __restrict__ hp) {
    const int lane = threadIdx.x;                       // 0..31
    const int m0   = blockIdx.x * 16;                   // row tile
    const int n0   = (blockIdx.y * blockDim.y + threadIdx.y) * 16;  // col tile (0..112)

    const int mrow   = m0 + (lane & 15);
    const int kbaseA = (lane < 16) ? 0 : 8;             // ISA 16-bit A layout
    const int ncol   = n0 + (lane & 15);
    const int kbaseB = (lane < 16) ? 0 : 8;             // ISA 16-bit B layout

    union Frag { v16bf v; u32x4 q[2]; };
    Frag a, b;
    v8f c = {};

    const __bf16* arow = hbf + (size_t)mrow * FDIM;
    const __bf16* brow = WT  + (size_t)ncol * FDIM;

    for (int k0 = 0; k0 < FDIM; k0 += 32) {
        // A: lanes hold K = kbaseA + {0..7, 16..23}
        const __bf16* pa = arow + k0 + kbaseA;
        a.q[0] = *(const u32x4*)(pa);
        a.q[1] = *(const u32x4*)(pa + 16);
        // B: lanes hold K = kbaseB + {0..7, 16..23} in transposed weight
        const __bf16* pb = brow + k0 + kbaseB;
        b.q[0] = *(const u32x4*)(pb);
        b.q[1] = *(const u32x4*)(pb + 16);
        c = __builtin_amdgcn_wmma_f32_16x16x32_bf16(false, a.v, false, b.v,
                                                    (short)0, c, false, false);
    }

    // D layout: VGPR r -> row (r | lane-hi*8), col = lane&15
    const int rbase = m0 + ((lane >= 16) ? 8 : 0);
    for (int r = 0; r < 8; ++r)
        hp[(size_t)(rbase + r) * FDIM + ncol] = c[r];
}

// ---------------- attention scalars per node ----------------
__global__ void k_asad(const float* __restrict__ hp, const float* __restrict__ att_src,
                       const float* __restrict__ att_dst, float* __restrict__ a_s,
                       float* __restrict__ a_d, int l) {
    int i = blockIdx.x * blockDim.x + threadIdx.x;
    if (i >= NN * NH) return;
    int n = i / NH, h = i % NH;
    const float* row = hp + (size_t)n * FDIM + h * NC;
    const float* as = att_src + (size_t)l * NH * NC + h * NC;
    const float* ad = att_dst + (size_t)l * NH * NC + h * NC;
    float s = 0.f, d = 0.f;
    for (int c = 0; c < NC; ++c) { float v = row[c]; s += v * as[c]; d += v * ad[c]; }
    a_s[i] = s; a_d[i] = d;
}

// ---------------- per-edge alpha + segment max ----------------
__global__ void k_edge_alpha(const int* __restrict__ ei, const float* __restrict__ eattr,
                             const float* __restrict__ loop_attr, const float* __restrict__ vedge_l,
                             const float* __restrict__ a_s, const float* __restrict__ a_d,
                             float* __restrict__ alpha, unsigned* __restrict__ mbuf) {
    int e = blockIdx.x * blockDim.x + threadIdx.x;
    if (e >= EF) return;
    int s, d;
    const float* ea;
    if (e < NE) { s = ei[e]; d = ei[NE + e]; ea = eattr + (size_t)e * NED; }
    else        { s = d = e - NE;            ea = loop_attr + (size_t)s * NED; }
    float ae[NH] = {0.f, 0.f, 0.f, 0.f};
    for (int j = 0; j < NED; ++j) {
        float v = ea[j];
        for (int h = 0; h < NH; ++h) ae[h] += v * vedge_l[j * NH + h];
    }
    for (int h = 0; h < NH; ++h) {
        float a = a_s[s * NH + h] + a_d[d * NH + h] + ae[h];
        a = (a > 0.f) ? a : NEG_SLOPE * a;                 // leaky_relu
        alpha[(size_t)e * NH + h] = a;
        atomicMax(mbuf + d * NH + h, fenc(a));
    }
}

// ---------------- fused exp + denom + unnormalized aggregation ----------------
__global__ void k_edge_agg(const int* __restrict__ ei, const float* __restrict__ alpha,
                           const unsigned* __restrict__ mbuf, const float* __restrict__ hp,
                           float* __restrict__ denom, float* __restrict__ agg) {
    int e = blockIdx.x * blockDim.x + threadIdx.x;
    if (e >= EF) return;
    int s, d;
    if (e < NE) { s = ei[e]; d = ei[NE + e]; }
    else        { s = d = e - NE; }
    const float* hrow = hp  + (size_t)s * FDIM;
    float*       arow = agg + (size_t)d * FDIM;
    for (int h = 0; h < NH; ++h) {
        float ex = __expf(alpha[(size_t)e * NH + h] - fdec(mbuf[d * NH + h]));
        atomicAdd(denom + d * NH + h, ex);
        const float* hs = hrow + h * NC;
        float*       ag = arow + h * NC;
        for (int c = 0; c < NC; ++c) atomicAdd(ag + c, ex * hs[c]);
    }
}

// ---------------- node update: normalize + bias + relu -> next-layer bf16 ----------------
__global__ void k_node_update(const float* __restrict__ agg, const float* __restrict__ denom,
                              const float* __restrict__ bias_l, __bf16* __restrict__ hbf,
                              float* __restrict__ hf, int last) {
    int i = blockIdx.x * blockDim.x + threadIdx.x;
    if (i >= NN * FDIM) return;
    int n = i / FDIM, j = i % FDIM;
    float v = agg[i] / denom[n * NH + j / NC] + bias_l[j];
    v = fmaxf(v, 0.f);
    hbf[i] = (__bf16)v;
    if (last) hf[i] = v;
}

// ---------------- mean pool (chunked partial sums, 1 atomic per chunk/col) ----------------
#define MEAN_CHUNK 256
__global__ void k_mean(const float* __restrict__ hf, float* __restrict__ g) {
    int i = blockIdx.x * blockDim.x + threadIdx.x;
    const int nchunks = (NN + MEAN_CHUNK - 1) / MEAN_CHUNK;
    if (i >= FDIM * nchunks) return;
    int col = i % FDIM, chunk = i / FDIM;
    int n0 = chunk * MEAN_CHUNK;
    int n1 = min(n0 + MEAN_CHUNK, NN);
    float s = 0.f;
    for (int n = n0; n < n1; ++n) s += hf[(size_t)n * FDIM + col];
    atomicAdd(g + col, s * (1.0f / (float)NN));
}

// ---------------- dense head: relu(g@W1+b1)@W2+b2 ----------------
__global__ void k_head(const float* __restrict__ g, const float* __restrict__ W1,
                       const float* __restrict__ b1, const float* __restrict__ W2,
                       const float* __restrict__ b2, float* __restrict__ out) {
    __shared__ float gg[FDIM];
    __shared__ float hid[2 * FDIM];
    int t = threadIdx.x;                     // 256 threads
    if (t < FDIM) gg[t] = g[t];
    __syncthreads();
    float s = b1[t];
    for (int k = 0; k < FDIM; ++k) s += gg[k] * W1[k * (2 * FDIM) + t];
    hid[t] = fmaxf(s, 0.f);
    __syncthreads();
    if (t < 2) {
        float o = b2[t];
        for (int j = 0; j < 2 * FDIM; ++j) o += hid[j] * W2[j * 2 + t];
        out[t] = o;
    }
}

// ---------------- host orchestration ----------------
static inline int cdiv(int a, int b) { return (a + b - 1) / b; }

extern "C" void kernel_launch(void* const* d_in, const int* in_sizes, int n_in,
                              void* d_out, int out_size, void* d_ws, size_t ws_size,
                              hipStream_t stream) {
    (void)in_sizes; (void)n_in; (void)out_size; (void)ws_size;
    const float* x         = (const float*)d_in[0];
    const int*   ei        = (const int*)  d_in[1];   // [2,E]
    const float* eattr     = (const float*)d_in[2];   // [E,16]
    const float* Wsrc      = (const float*)d_in[3];   // [3,128,128]
    const float* att_src   = (const float*)d_in[4];   // [3,4,32]
    const float* att_dst   = (const float*)d_in[5];
    const float* Wedge     = (const float*)d_in[6];   // [3,16,128]
    const float* att_edge  = (const float*)d_in[7];
    const float* bias      = (const float*)d_in[8];   // [3,128]
    const float* W1        = (const float*)d_in[9];   // [128,256]
    const float* b1        = (const float*)d_in[10];
    const float* W2        = (const float*)d_in[11];  // [256,2]
    const float* b2        = (const float*)d_in[12];
    float* out = (float*)d_out;

    // workspace carving (256B aligned)
    size_t off = 0;
    char* base = (char*)d_ws;
    auto carve = [&](size_t bytes) -> char* {
        char* p = base + off;
        off += (bytes + 255) & ~(size_t)255;
        return p;
    };
    __bf16*   hbf       = (__bf16*)  carve((size_t)NN * FDIM * 2);
    __bf16*   WT        = (__bf16*)  carve((size_t)NL * FDIM * FDIM * 2);
    float*    vedge     = (float*)   carve((size_t)NL * NED * NH * 4);
    float*    deg       = (float*)   carve((size_t)NN * 4);
    float*    loopsum   = (float*)   carve((size_t)NN * NED * 4);
    float*    loop_attr = (float*)   carve((size_t)NN * NED * 4);
    float*    hp        = (float*)   carve((size_t)NN * FDIM * 4);
    float*    a_s       = (float*)   carve((size_t)NN * NH * 4);
    float*    a_d       = (float*)   carve((size_t)NN * NH * 4);
    float*    alpha     = (float*)   carve((size_t)EF * NH * 4);
    unsigned* mbuf      = (unsigned*)carve((size_t)NN * NH * 4);
    float*    denom     = (float*)   carve((size_t)NN * NH * 4);
    float*    agg       = (float*)   carve((size_t)NN * FDIM * 4);
    float*    hf        = (float*)   carve((size_t)NN * FDIM * 4);
    float*    g         = (float*)   carve((size_t)FDIM * 4);

    const int B = 256;

    // --- prep (layer independent) ---
    k_x_to_bf<<<cdiv(NN * FDIM, B), B, 0, stream>>>(x, hbf, NN * FDIM);
    k_wt<<<cdiv(NL * FDIM * FDIM, B), B, 0, stream>>>(Wsrc, WT);
    k_vedge<<<cdiv(NL * NED * NH, B), B, 0, stream>>>(Wedge, att_edge, vedge);
    k_fill_f32<<<cdiv(NN, B), B, 0, stream>>>(deg, 0.f, NN);
    k_fill_f32<<<cdiv(NN * NED, B), B, 0, stream>>>(loopsum, 0.f, NN * NED);
    k_deg_loop<<<cdiv(NE, B), B, 0, stream>>>(ei, eattr, deg, loopsum);
    k_loop_div<<<cdiv(NN * NED, B), B, 0, stream>>>(loopsum, deg, loop_attr);

    // --- GAT layers ---
    for (int l = 0; l < NL; ++l) {
        k_fill_f32<<<cdiv(NN * NH, B), B, 0, stream>>>((float*)mbuf, 0.f, NN * NH); // enc(-inf) lower bound
        k_fill_f32<<<cdiv(NN * NH, B), B, 0, stream>>>(denom, 0.f, NN * NH);
        k_fill_f32<<<cdiv(NN * FDIM, B), B, 0, stream>>>(agg, 0.f, NN * FDIM);

        dim3 ggrid(NN / 16, 2), gblk(32, 4);
        k_gemm_hp<<<ggrid, gblk, 0, stream>>>(hbf, WT + (size_t)l * FDIM * FDIM, hp);

        k_asad<<<cdiv(NN * NH, B), B, 0, stream>>>(hp, att_src, att_dst, a_s, a_d, l);
        k_edge_alpha<<<cdiv(EF, B), B, 0, stream>>>(ei, eattr, loop_attr,
                                                    vedge + (size_t)l * NED * NH,
                                                    a_s, a_d, alpha, mbuf);
        k_edge_agg<<<cdiv(EF, B), B, 0, stream>>>(ei, alpha, mbuf, hp, denom, agg);
        k_node_update<<<cdiv(NN * FDIM, B), B, 0, stream>>>(agg, denom, bias + (size_t)l * FDIM,
                                                            hbf, hf, (l == NL - 1) ? 1 : 0);
    }

    // --- mean pool + MLP head ---
    k_fill_f32<<<1, B, 0, stream>>>(g, 0.f, FDIM);
    const int nchunks = (NN + MEAN_CHUNK - 1) / MEAN_CHUNK;
    k_mean<<<cdiv(FDIM * nchunks, B), B, 0, stream>>>(hf, g);
    k_head<<<1, 2 * FDIM, 0, stream>>>(g, W1, b1, W2, b2, out);
}
